// SwinWindowAttention_27685359190424
// MI455X (gfx1250) — matrix operand order repaired and checked by reading, not compiled, not run on hardware
//
#include <hip/hip_runtime.h>
#include <hip/hip_bf16.h>

typedef _Float16 half_t;
typedef __attribute__((ext_vector_type(16))) _Float16 v16h;
typedef __attribute__((ext_vector_type(8)))  _Float16 v8h;
typedef __attribute__((ext_vector_type(4)))  _Float16 v4h;
typedef __attribute__((ext_vector_type(8)))  float    v8f;
typedef __attribute__((ext_vector_type(4)))  float    vf4;

// global (addrspace 1) views to force global_load_* instead of flat_load_*
typedef const __attribute__((address_space(1))) v8h gv8h;
typedef const __attribute__((address_space(1))) vf4 gvf4;
typedef       __attribute__((address_space(1))) vf4 gvf4_mut;

#define WSZ     7
#define SHIFT_  3
#define HWDIM   112
#define NWIN    16
#define CH      256
#define NHEADS  8
#define HD      32
#define TOK     49
#define TOKP    64
#define STR_WIN 264       // half stride for win / Q / ctx
#define STR_P   72        // half stride for probs
#define LN_EPS  0.001f

// ---------------- WMMA fragment helpers (per ISA 05_wmma layouts) -----------
__device__ __forceinline__ v16h load_a_frag(const half_t* base, int m0, int k0,
                                            int stride, int lane) {
  int row = m0 + (lane & 15);
  int kb  = k0 + ((lane >> 4) << 3);
  const half_t* p = base + row * stride + kb;
  v8h lo = *(const v8h*)(p);
  v8h hi = *(const v8h*)(p + 16);
  return __builtin_shufflevector(lo, hi, 0,1,2,3,4,5,6,7,8,9,10,11,12,13,14,15);
}

__device__ __forceinline__ v16h load_b_frag(const half_t* base, int k0, int n0,
                                            int stride, int lane) {
  const half_t* p = base + (size_t)(k0 + ((lane >> 4) << 3)) * stride + n0 + (lane & 15);
  v16h r;
#pragma unroll
  for (int i = 0; i < 8; ++i) {
    r[i]     = p[(size_t)i * stride];
    r[8 + i] = p[(size_t)(16 + i) * stride];
  }
  return r;
}

__device__ __forceinline__ v16h load_b_frag_g(const half_t* base, int k0, int n0,
                                              int stride, int lane) {
  const half_t* p = base + (size_t)(k0 + ((lane >> 4) << 3)) * stride + n0 + (lane & 15);
  v16h r;
#pragma unroll
  for (int i = 0; i < 8; ++i) {
    r[i]     = p[(size_t)i * stride];
    r[8 + i] = p[(size_t)(16 + i) * stride];
  }
  return r;
}

__device__ __forceinline__ v8f wmma_f16(v16h a, v16h b, v8f c) {
  return __builtin_amdgcn_wmma_f32_16x16x32_f16(false, a, false, b,
                                                (short)0, c, false, false);
}

// ---------------- weight fp32 -> f16 pre-conversion -------------------------
__global__ void convert_weights_kernel(const float* __restrict__ wq,
                                       const float* __restrict__ wk,
                                       const float* __restrict__ wv,
                                       const float* __restrict__ wo,
                                       half_t* __restrict__ dst) {
  int i = blockIdx.x * blockDim.x + threadIdx.x;
  if (i < CH * CH) {
    dst[i]            = (half_t)wq[i];
    dst[CH*CH   + i]  = (half_t)wk[i];
    dst[2*CH*CH + i]  = (half_t)wv[i];
    dst[3*CH*CH + i]  = (half_t)wo[i];
  }
}

// ---------------- main fused window-attention kernel ------------------------
constexpr size_t SMEM_BYTES =
    (size_t)(TOKP*STR_WIN + TOKP*STR_WIN + CH*TOKP + TOKP*CH +
             NHEADS*TOKP*STR_P) * 2 + (size_t)TOKP*CH*4;   // = 272384 B

__launch_bounds__(256)
__global__ void swin_window_attn_kernel(const float* __restrict__ x,
                                        const half_t* __restrict__ wH,
                                        const float* __restrict__ bq,
                                        const float* __restrict__ bk,
                                        const float* __restrict__ bv,
                                        const float* __restrict__ bo,
                                        const float* __restrict__ gamma,
                                        const float* __restrict__ beta,
                                        float* __restrict__ out) {
  extern __shared__ __align__(16) char smem_raw[];
  half_t* sWin = (half_t*)smem_raw;
  half_t* sQ   = sWin + TOKP*STR_WIN;        // later reused as ctx
  half_t* sKt  = sQ   + TOKP*STR_WIN;        // [channel][token]
  half_t* sV   = sKt  + CH*TOKP;
  half_t* sP   = sV   + TOKP*CH;
  float*  sY   = (float*)(sP + NHEADS*TOKP*STR_P);

  const int tid  = threadIdx.x;
  const int wave = tid >> 5;
  const int lane = tid & 31;

  const int win = blockIdx.x;
  const int b   = win >> 8;
  const int wh  = (win >> 4) & 15;
  const int ww  = win & 15;

  // ---- Phase 0: gather shifted window into LDS (f16), zero pad rows --------
  for (int s = tid >> 3; s < TOK; s += 32) {
    int i = s / WSZ, j = s % WSZ;
    int hh  = (wh * WSZ + i + SHIFT_) % HWDIM;
    int ww2 = (ww * WSZ + j + SHIFT_) % HWDIM;
    const gvf4* src = (const gvf4*)(x +
        (((size_t)b * HWDIM + hh) * HWDIM + ww2) * CH + (tid & 7) * 32);
    half_t* dst = sWin + s * STR_WIN + (tid & 7) * 32;
#pragma unroll
    for (int c = 0; c < 8; ++c) {
      vf4 v = src[c];
      v4h h4 = { (half_t)v.x, (half_t)v.y, (half_t)v.z, (half_t)v.w };
      *(v4h*)(dst + c * 4) = h4;
    }
  }
  {
    v4h z = {};
    for (int idx = tid; idx < (TOKP - TOK) * (CH / 4); idx += 256) {
      int r = TOK + idx / (CH / 4), c = (idx % (CH / 4)) * 4;
      *(v4h*)(sWin + r * STR_WIN + c) = z;
    }
  }
  __syncthreads();

  // ---- Phase 1: Q,K,V projections — 3 WMMA chains, A-frags hoisted ---------
  {
    const half_t* Wq = wH;
    const half_t* Wk = wH + CH*CH;
    const half_t* Wv = wH + 2*CH*CH;
#pragma unroll 1
    for (int mt = 0; mt < 4; ++mt) {
      v16h a[8];
#pragma unroll
      for (int kc = 0; kc < 8; ++kc)
        a[kc] = load_a_frag(sWin, mt*16, kc*32, STR_WIN, lane);
#pragma unroll 1
      for (int hlf = 0; hlf < 2; ++hlf) {
        const int nt = wave + 8*hlf;
        const int n  = nt*16 + (lane & 15);
        v8f accq = {}, acck = {}, accv = {};
#pragma unroll
        for (int kc = 0; kc < 8; ++kc) {
          accq = wmma_f16(a[kc], load_b_frag_g(Wq, kc*32, nt*16, CH, lane), accq);
          acck = wmma_f16(a[kc], load_b_frag_g(Wk, kc*32, nt*16, CH, lane), acck);
          accv = wmma_f16(a[kc], load_b_frag_g(Wv, kc*32, nt*16, CH, lane), accv);
        }
        const float bqv = bq[n], bkv = bk[n], bvv = bv[n];
        const int mbase = mt*16 + ((lane >> 4) << 3);
#pragma unroll
        for (int r = 0; r < 8; ++r) {
          int m = mbase + r;
          sQ [m * STR_WIN + n] = (half_t)(accq[r] + bqv);
          sKt[n * TOKP + m]    = (half_t)(acck[r] + bkv);   // transposed
          sV [m * CH + n]      = (half_t)(accv[r] + bvv);
        }
      }
    }
  }
  __syncthreads();

  // ---- Phase 2: per-head scores + streaming softmax (per 16-row block) -----
  {
    const int h = wave;
    v16h bkf[4];
#pragma unroll
    for (int nt = 0; nt < 4; ++nt)
      bkf[nt] = load_b_frag(sKt, h*HD, nt*16, TOKP, lane);

    const float scale = 0.17677669529663687f;   // 1/sqrt(32)
    const int colb    = lane & 15;
    half_t* pbuf      = sP + h * TOKP * STR_P;

#pragma unroll 1
    for (int mt = 0; mt < 4; ++mt) {
      v16h aq = load_a_frag(sQ, mt*16, h*HD, STR_WIN, lane);
      v8f sc[4];
#pragma unroll
      for (int nt = 0; nt < 4; ++nt) {
        v8f z = {};
        sc[nt] = wmma_f16(aq, bkf[nt], z);
      }
#pragma unroll
      for (int r = 0; r < 8; ++r) {
        float vals[4];
        float mx = -3.0e38f;
#pragma unroll
        for (int nt = 0; nt < 4; ++nt) {
          int col = nt*16 + colb;
          float v = (col < TOK) ? (float)sc[nt][r] * scale : -3.0e38f;
          vals[nt] = v;
          mx = fmaxf(mx, v);
        }
#pragma unroll
        for (int off = 1; off < 16; off <<= 1)
          mx = fmaxf(mx, __shfl_xor(mx, off, 32));
        float sum = 0.f;
#pragma unroll
        for (int nt = 0; nt < 4; ++nt) {
          float e = __expf(vals[nt] - mx);
          vals[nt] = e; sum += e;
        }
#pragma unroll
        for (int off = 1; off < 16; off <<= 1)
          sum += __shfl_xor(sum, off, 32);
        float inv = 1.0f / sum;
        int m = mt*16 + ((lane >> 4) << 3) + r;
#pragma unroll
        for (int nt = 0; nt < 4; ++nt)
          pbuf[m * STR_P + nt*16 + colb] = (half_t)(vals[nt] * inv);
      }
    }

    // ---- Phase 3: ctx = probs @ V, B-fragments hoisted across M-tiles ------
    v16h bvf[2][2];
#pragma unroll
    for (int kc = 0; kc < 2; ++kc)
#pragma unroll
      for (int ntd = 0; ntd < 2; ++ntd)
        bvf[kc][ntd] = load_b_frag(sV, kc*32, h*HD + ntd*16, CH, lane);
#pragma unroll
    for (int mt = 0; mt < 4; ++mt) {
      v16h a0 = load_a_frag(pbuf, mt*16, 0,  STR_P, lane);
      v16h a1 = load_a_frag(pbuf, mt*16, 32, STR_P, lane);
      v8f c0 = {}, c1 = {};
      c0 = wmma_f16(a0, bvf[0][0], c0);  c1 = wmma_f16(a0, bvf[0][1], c1);
      c0 = wmma_f16(a1, bvf[1][0], c0);  c1 = wmma_f16(a1, bvf[1][1], c1);
      int mbase = mt*16 + ((lane >> 4) << 3);
      int nA = h*HD + (lane & 15), nB = nA + 16;
#pragma unroll
      for (int r = 0; r < 8; ++r) {
        sQ[(mbase + r) * STR_WIN + nA] = (half_t)c0[r];
        sQ[(mbase + r) * STR_WIN + nB] = (half_t)c1[r];
      }
    }
  }
  __syncthreads();

  // ---- Phase 4: output projection + bias + residual -> sY (f32) -----------
  {
    const half_t* Wo = wH + 3*CH*CH;
    const int nt0 = wave, nt1 = wave + 8;
    const int n0  = nt0*16 + (lane & 15);
    const int n1  = nt1*16 + (lane & 15);
    const float bo0 = bo[n0], bo1 = bo[n1];
#pragma unroll 1
    for (int mt = 0; mt < 4; ++mt) {
      v16h a[8];
#pragma unroll
      for (int kc = 0; kc < 8; ++kc)
        a[kc] = load_a_frag(sQ, mt*16, kc*32, STR_WIN, lane);
      v8f c0 = {}, c1 = {};
#pragma unroll
      for (int kc = 0; kc < 8; ++kc) {
        c0 = wmma_f16(a[kc], load_b_frag_g(Wo, kc*32, nt0*16, CH, lane), c0);
        c1 = wmma_f16(a[kc], load_b_frag_g(Wo, kc*32, nt1*16, CH, lane), c1);
      }
      const int mbase = mt*16 + ((lane >> 4) << 3);
#pragma unroll
      for (int r = 0; r < 8; ++r) {
        int m = mbase + r;
        sY[m * CH + n0] = c0[r] + bo0 + (float)sWin[m * STR_WIN + n0];
        sY[m * CH + n1] = c1[r] + bo1 + (float)sWin[m * STR_WIN + n1];
      }
    }
  }
  __syncthreads();

  // ---- Phase 5: LayerNorm + inverse-roll scatter ---------------------------
  for (int s = wave; s < TOK; s += 8) {
    const vf4* yr = (const vf4*)(sY + s * CH + lane * 8);
    vf4 y0 = yr[0];
    vf4 y1 = yr[1];
    float sum = y0.x + y0.y + y0.z + y0.w + y1.x + y1.y + y1.z + y1.w;
    float ssq = y0.x*y0.x + y0.y*y0.y + y0.z*y0.z + y0.w*y0.w
              + y1.x*y1.x + y1.y*y1.y + y1.z*y1.z + y1.w*y1.w;
#pragma unroll
    for (int off = 1; off < 32; off <<= 1) {
      sum += __shfl_xor(sum, off, 32);
      ssq += __shfl_xor(ssq, off, 32);
    }
    float mean = sum * (1.0f / CH);
    float var  = ssq * (1.0f / CH) - mean * mean;
    float rstd = rsqrtf(var + LN_EPS);

    int i = s / WSZ, j = s % WSZ;
    int hh  = (wh * WSZ + i + SHIFT_) % HWDIM;
    int ww2 = (ww * WSZ + j + SHIFT_) % HWDIM;
    gvf4_mut* dst = (gvf4_mut*)(out +
        (((size_t)b * HWDIM + hh) * HWDIM + ww2) * CH + lane * 8);

    const gvf4* gg = (const gvf4*)(gamma + lane*8);
    const gvf4* bb = (const gvf4*)(beta  + lane*8);
    vf4 g0 = gg[0], g1 = gg[1];
    vf4 b0 = bb[0], b1 = bb[1];
    vf4 o0, o1;
    o0.x = (y0.x-mean)*rstd*g0.x + b0.x;  o0.y = (y0.y-mean)*rstd*g0.y + b0.y;
    o0.z = (y0.z-mean)*rstd*g0.z + b0.z;  o0.w = (y0.w-mean)*rstd*g0.w + b0.w;
    o1.x = (y1.x-mean)*rstd*g1.x + b1.x;  o1.y = (y1.y-mean)*rstd*g1.y + b1.y;
    o1.z = (y1.z-mean)*rstd*g1.z + b1.z;  o1.w = (y1.w-mean)*rstd*g1.w + b1.w;
    dst[0] = o0;
    dst[1] = o1;
  }
}

// ---------------------------------------------------------------------------
extern "C" void kernel_launch(void* const* d_in, const int* in_sizes, int n_in,
                              void* d_out, int out_size, void* d_ws, size_t ws_size,
                              hipStream_t stream) {
  (void)in_sizes; (void)n_in; (void)out_size; (void)ws_size;
  const float* x     = (const float*)d_in[0];
  const float* wq    = (const float*)d_in[1];
  const float* bq    = (const float*)d_in[2];
  const float* wk    = (const float*)d_in[3];
  const float* bk    = (const float*)d_in[4];
  const float* wv    = (const float*)d_in[5];
  const float* bv    = (const float*)d_in[6];
  const float* wo    = (const float*)d_in[7];
  const float* bo    = (const float*)d_in[8];
  const float* gamma = (const float*)d_in[9];
  const float* beta  = (const float*)d_in[10];
  float* out = (float*)d_out;
  half_t* wH = (half_t*)d_ws;     // 4 x 256x256 f16 = 512 KB scratch

  convert_weights_kernel<<<CH*CH/256, 256, 0, stream>>>(wq, wk, wv, wo, wH);

  const int nwin = 16 * NWIN * NWIN;   // 4096 windows
  swin_window_attn_kernel<<<nwin, 256, SMEM_BYTES, stream>>>(
      x, wH, bq, bk, bv, bo, gamma, beta, out);
}
